// DualOutputMoE_21620865368076
// MI455X (gfx1250) — matrix operand order repaired and checked by reading, not compiled, not run on hardware
//
#include <hip/hip_runtime.h>
#include <stdint.h>

#define T_TOK 8192
#define H_DIM 2048
#define E_NUM 8

typedef __attribute__((ext_vector_type(16))) __bf16 v16bf;
typedef __attribute__((ext_vector_type(8)))  float  v8f;

__device__ __forceinline__ uint16_t f32_to_bf16(float f) {
    union { float f; uint32_t u; } c; c.f = f;
    uint32_t u = c.u;
    uint32_t r = u + 0x7FFFu + ((u >> 16) & 1u); // round-to-nearest-even
    return (uint16_t)(r >> 16);
}

// ---------------------------------------------------------------------------
// Kernel A: routing + x fp32->bf16 conversion.
// One wave32 per token. Wg is [H, E=8] row-major: 8 contiguous floats per h.
// ---------------------------------------------------------------------------
__global__ void moe_route_convert(const float* __restrict__ x,
                                  const float* __restrict__ Wg,
                                  const float* __restrict__ bg,
                                  uint16_t* __restrict__ xb,
                                  int*   __restrict__ cnt,
                                  int*   __restrict__ idxl,
                                  float* __restrict__ gatel) {
    const int wave = threadIdx.x >> 5;
    const int lane = threadIdx.x & 31;
    const int t = blockIdx.x * 8 + wave;

    const float* xr  = x  + (size_t)t * H_DIM;
    uint16_t*    xbr = xb + (size_t)t * H_DIM;

    float acc[E_NUM];
#pragma unroll
    for (int e = 0; e < E_NUM; ++e) acc[e] = 0.f;

    for (int h = lane; h < H_DIM; h += 32) {
        float xv = xr[h];
        xbr[h] = f32_to_bf16(xv);
        const float4* wg = (const float4*)(Wg + (size_t)h * E_NUM);
        float4 w0 = wg[0], w1 = wg[1];
        acc[0] += xv * w0.x; acc[1] += xv * w0.y;
        acc[2] += xv * w0.z; acc[3] += xv * w0.w;
        acc[4] += xv * w1.x; acc[5] += xv * w1.y;
        acc[6] += xv * w1.z; acc[7] += xv * w1.w;
    }
    // full-wave xor reduction (wave32)
#pragma unroll
    for (int e = 0; e < E_NUM; ++e) {
#pragma unroll
        for (int off = 16; off > 0; off >>= 1)
            acc[e] += __shfl_xor(acc[e], off, 32);
    }

    if (lane == 0) {
        float lg[E_NUM], p[E_NUM];
        float mx = -1e30f;
#pragma unroll
        for (int e = 0; e < E_NUM; ++e) { lg[e] = acc[e] + bg[e]; mx = fmaxf(mx, lg[e]); }
        float sum = 0.f;
#pragma unroll
        for (int e = 0; e < E_NUM; ++e) { p[e] = __expf(lg[e] - mx); sum += p[e]; }
        const float inv = 1.f / sum;
        int i0 = 0;
#pragma unroll
        for (int e = 1; e < E_NUM; ++e) if (p[e] > p[i0]) i0 = e;
        int i1 = (i0 == 0) ? 1 : 0;
#pragma unroll
        for (int e = 0; e < E_NUM; ++e) if (e != i0 && p[e] > p[i1]) i1 = e;

        int p0 = atomicAdd(&cnt[i0], 1);
        idxl [i0 * T_TOK + p0] = t;
        gatel[i0 * T_TOK + p0] = p[i0] * inv;
        int p1 = atomicAdd(&cnt[i1], 1);
        idxl [i1 * T_TOK + p1] = t;
        gatel[i1 * T_TOK + p1] = p[i1] * inv;
    }
}

// ---------------------------------------------------------------------------
// Kernel B: We [E][K][N] fp32 -> Wt [E][N][K] bf16 (transpose + convert).
// 32x32 LDS tiles; Wt (67 MB) + xb (33.5 MB) then live in the 192 MB L2
// for the GEMM pass, so per-expert weight re-reads never touch HBM.
// ---------------------------------------------------------------------------
__global__ void wet_transpose(const float* __restrict__ We,
                              uint16_t* __restrict__ Wt) {
    __shared__ float tile[32][33];
    const int e  = blockIdx.z;
    const int k0 = blockIdx.y * 32;
    const int n0 = blockIdx.x * 32;
    const float* src = We + ((size_t)e * H_DIM + k0) * H_DIM + n0;
    for (int r = threadIdx.y; r < 32; r += 8)
        tile[r][threadIdx.x] = src[(size_t)r * H_DIM + threadIdx.x];
    __syncthreads();
    uint16_t* dst = Wt + ((size_t)e * H_DIM + n0) * H_DIM + k0;
    for (int r = threadIdx.y; r < 32; r += 8)
        dst[(size_t)r * H_DIM + threadIdx.x] = f32_to_bf16(tile[threadIdx.x][r]);
}

// ---------------------------------------------------------------------------
// Kernel C: per-expert gathered GEMM with v_wmma_f32_16x16x32_bf16.
// Block = 256 threads = 8 waves; block tile = 16 tokens x 512 cols.
// Each wave owns a 16x64 tile (4 accumulators): one A-load pair feeds
// 4 WMMAs -> 2.5 b128 loads per WMMA instead of 4.
// Scatter with gate weighting via hardware f32 atomic add (top-2 overlap).
// ---------------------------------------------------------------------------
__global__ void moe_expert_gemm(const uint16_t* __restrict__ xb,
                                const uint16_t* __restrict__ Wt,
                                const float* __restrict__ be,
                                const int*   __restrict__ cnt,
                                const int*   __restrict__ idxl,
                                const float* __restrict__ gatel,
                                float* __restrict__ out) {
    const int e     = blockIdx.z;
    const int tile0 = blockIdx.y * 16;
    const int n0    = blockIdx.x * 512;
    const int count = cnt[e];
    if (tile0 >= count) return;

    __shared__ int   s_tok[16];
    __shared__ float s_gate[16];
    if (threadIdx.x < 16) {
        int j = tile0 + (int)threadIdx.x;
        if (j < count) {
            s_tok[threadIdx.x]  = idxl [e * T_TOK + j];
            s_gate[threadIdx.x] = gatel[e * T_TOK + j];
        } else {
            s_tok[threadIdx.x]  = 0;
            s_gate[threadIdx.x] = 0.f;   // padded rows contribute zero
        }
    }
    __syncthreads();

    const int wave  = threadIdx.x >> 5;
    const int lane  = threadIdx.x & 31;
    const int half  = lane >> 4;      // 0: lanes 0-15, 1: lanes 16-31
    const int l15   = lane & 15;
    const int nbase = n0 + wave * 64 + l15;   // 4 subtiles: nbase + 16*j

    // A (16x32 bf16): lanes 0-15 row M=l15 hold K {0..7,16..23};
    //                 lanes 16-31 same rows hold K {8..15,24..31}.
    const uint16_t* arow = xb + (size_t)s_tok[l15] * H_DIM + (half ? 8 : 0);
    // B (32x16 bf16): lanes 0-15 col N hold K {0..7,16..23}; lanes 16-31 K {8..15,24..31}.
    // Subtile j's column is nbase + 16*j -> constant 16*H element stride,
    // foldable into the instruction's immediate offset.
    const uint16_t* bcol = Wt + ((size_t)e * H_DIM + nbase) * H_DIM + (half ? 8 : 0);

    v8f c0 = {}, c1 = {}, c2 = {}, c3 = {};
#pragma unroll 2
    for (int k0 = 0; k0 < H_DIM; k0 += 32) {
        union { uint4 u[2]; v16bf v; } A, B0, B1, B2, B3;
        A.u[0]  = *(const uint4*)(arow + k0);        // K k0+{0..7}/{8..15}
        A.u[1]  = *(const uint4*)(arow + k0 + 16);   // K k0+{16..23}/{24..31}
        B0.u[0] = *(const uint4*)(bcol + k0);
        B0.u[1] = *(const uint4*)(bcol + k0 + 16);
        B1.u[0] = *(const uint4*)(bcol + 16 * H_DIM + k0);
        B1.u[1] = *(const uint4*)(bcol + 16 * H_DIM + k0 + 16);
        B2.u[0] = *(const uint4*)(bcol + 32 * H_DIM + k0);
        B2.u[1] = *(const uint4*)(bcol + 32 * H_DIM + k0 + 16);
        B3.u[0] = *(const uint4*)(bcol + 48 * H_DIM + k0);
        B3.u[1] = *(const uint4*)(bcol + 48 * H_DIM + k0 + 16);
        c0 = __builtin_amdgcn_wmma_f32_16x16x32_bf16(false, A.v, false, B0.v, (short)0, c0, false, false);
        c1 = __builtin_amdgcn_wmma_f32_16x16x32_bf16(false, A.v, false, B1.v, (short)0, c1, false, false);
        c2 = __builtin_amdgcn_wmma_f32_16x16x32_bf16(false, A.v, false, B2.v, (short)0, c2, false, false);
        c3 = __builtin_amdgcn_wmma_f32_16x16x32_bf16(false, A.v, false, B3.v, (short)0, c3, false, false);
    }

    // C/D layout: VGPR r -> M = r + 8*half, N = l15 (per 16-col subtile).
    float gr[8];
#pragma unroll
    for (int r = 0; r < 8; ++r) gr[r] = s_gate[r + 8 * half];

#pragma unroll
    for (int j = 0; j < 4; ++j) {
        const v8f   c  = (j == 0) ? c0 : (j == 1) ? c1 : (j == 2) ? c2 : c3;
        const int   n  = nbase + 16 * j;
        const float bn = be[(size_t)e * H_DIM + n];
#pragma unroll
        for (int r = 0; r < 8; ++r) {
            const int   mrow = r + 8 * half;
            const float v    = gr[r] * (c[r] + bn);
            unsafeAtomicAdd(out + (size_t)s_tok[mrow] * H_DIM + n, v);
        }
    }
}

// ---------------------------------------------------------------------------
extern "C" void kernel_launch(void* const* d_in, const int* in_sizes, int n_in,
                              void* d_out, int out_size, void* d_ws, size_t ws_size,
                              hipStream_t stream) {
    const float* x  = (const float*)d_in[0];
    const float* Wg = (const float*)d_in[1];
    const float* bg = (const float*)d_in[2];
    const float* We = (const float*)d_in[3];
    const float* be = (const float*)d_in[4];
    float* out = (float*)d_out;

    char* ws = (char*)d_ws;
    size_t off = 0;
    uint16_t* Wt    = (uint16_t*)(ws + off); off += (size_t)E_NUM * H_DIM * H_DIM * 2; // 67 MB
    uint16_t* xb    = (uint16_t*)(ws + off); off += (size_t)T_TOK * H_DIM * 2;         // 33.5 MB
    int*      idxl  = (int*)     (ws + off); off += (size_t)E_NUM * T_TOK * 4;         // 256 KB
    float*    gatel = (float*)   (ws + off); off += (size_t)E_NUM * T_TOK * 4;         // 256 KB
    int*      cnt   = (int*)     (ws + off); off += 256;

    hipMemsetAsync(cnt, 0, E_NUM * sizeof(int), stream);
    hipMemsetAsync(d_out, 0, (size_t)out_size * sizeof(float), stream);

    moe_route_convert<<<T_TOK / 8, 256, 0, stream>>>(x, Wg, bg, xb, cnt, idxl, gatel);

    wet_transpose<<<dim3(H_DIM / 32, H_DIM / 32, E_NUM), dim3(32, 8), 0, stream>>>(We, Wt);

    moe_expert_gemm<<<dim3(H_DIM / 512, T_TOK / 16, E_NUM), 256, 0, stream>>>(
        xb, Wt, be, cnt, idxl, gatel, out);
}
